// Net_36361193128584
// MI455X (gfx1250) — compile-verified, matrix operand order checked by reading, not run on hardware
//
#include <hip/hip_runtime.h>
#include <hip/hip_bf16.h>

typedef __attribute__((ext_vector_type(16))) __bf16 v16bf;
typedef __attribute__((ext_vector_type(8)))  float  v8f;
typedef __attribute__((ext_vector_type(4)))  float  v4f;

#define N_NODES 65536
#define N_FEAT  256
#define N_EDGES 2097152
#define C_CH    16

// ---------------------------------------------------------------------------
// Weight pre-pack: f32 [k_rows x 16] row-major -> bf16 in WMMA B lane layout.
// Packed layout: [kblock][lane][elem], elem e<8 -> K = kb+e, e>=8 -> K = kb+8+e
// (kb = (lane>>4)*8), N = lane&15.  Out-of-range K (layers 2/3) packs as zero,
// so the K>=16 half of the 16x16x32 WMMA contributes nothing.
// ---------------------------------------------------------------------------
__global__ __launch_bounds__(256) void pack_w_kernel(
    const float* __restrict__ w, __bf16* __restrict__ pw, int k_rows, int nblocks)
{
    const int idx = blockIdx.x * 256 + threadIdx.x;
    if (idx >= nblocks * 512) return;
    const int b  = idx >> 9;
    const int L  = (idx >> 4) & 31;
    const int e  = idx & 15;
    const int n  = L & 15;
    const int kb = (L >> 4) * 8;
    const int k  = b * 32 + kb + (e < 8 ? e : 8 + e);
    const float v = (k < k_rows) ? w[k * C_CH + n] : 0.0f;
    pw[idx] = (__bf16)v;
}

// ---------------------------------------------------------------------------
// Layer-1 projection: msg = x @ w_rel, root = x @ w_root   (256 -> 16)
// One wave per 16-row tile; K=256 in 8 WMMA steps.  A fed by nontemporal
// b128 loads; B fed by one contiguous 32B v16bf load per matrix.
// ---------------------------------------------------------------------------
__global__ __launch_bounds__(32) void proj1_kernel(
    const float* __restrict__ x,
    const __bf16* __restrict__ pw_rel, const __bf16* __restrict__ pw_root,
    float* __restrict__ msg, float* __restrict__ root)
{
    const int lane = threadIdx.x;
    const int row0 = blockIdx.x * 16;
    const int m    = lane & 15;
    const int half = lane >> 4;
    const int kb   = half * 8;
    const int n    = lane & 15;

    const v4f*   xv  = (const v4f*)(x + (size_t)(row0 + m) * N_FEAT);
    const v16bf* wrv = (const v16bf*)pw_rel;
    const v16bf* wsv = (const v16bf*)pw_root;

    v8f acc_r = {};
    v8f acc_s = {};

    for (int kk = 0; kk < N_FEAT; kk += 32) {
        const int base = (kk + kb) >> 2;
        const v4f p0 = __builtin_nontemporal_load(&xv[base]);
        const v4f p1 = __builtin_nontemporal_load(&xv[base + 1]);
        const v4f q0 = __builtin_nontemporal_load(&xv[base + 4]);
        const v4f q1 = __builtin_nontemporal_load(&xv[base + 5]);
        v16bf a;
#pragma unroll
        for (int e = 0; e < 4; ++e) {
            a[e]      = (__bf16)p0[e];
            a[e + 4]  = (__bf16)p1[e];
            a[e + 8]  = (__bf16)q0[e];
            a[e + 12] = (__bf16)q1[e];
        }

        const v16bf br = wrv[(kk >> 5) * 32 + lane];
        const v16bf bs = wsv[(kk >> 5) * 32 + lane];

        acc_r = __builtin_amdgcn_wmma_f32_16x16x32_bf16(
            false, a, false, br, (short)0, acc_r, false, false);
        acc_s = __builtin_amdgcn_wmma_f32_16x16x32_bf16(
            false, a, false, bs, (short)0, acc_s, false, false);
    }

#pragma unroll
    for (int j = 0; j < 8; ++j) {
        const int row = row0 + j + half * 8;
        msg [row * C_CH + n] = acc_r[j];
        root[row * C_CH + n] = acc_s[j];
    }
}

// ---------------------------------------------------------------------------
// Layers 2/3 projection: 16 -> 16 GEMM; upper K half zero (packed as zeros).
// ---------------------------------------------------------------------------
__global__ __launch_bounds__(32) void projc_kernel(
    const float* __restrict__ hin,
    const __bf16* __restrict__ pw_rel, const __bf16* __restrict__ pw_root,
    float* __restrict__ msg, float* __restrict__ root)
{
    const int lane = threadIdx.x;
    const int row0 = blockIdx.x * 16;
    const int m    = lane & 15;
    const int half = lane >> 4;
    const int kb   = half * 8;
    const int n    = lane & 15;

    const v4f* hv = (const v4f*)(hin + (size_t)(row0 + m) * C_CH);
    const v4f p0 = hv[kb >> 2];
    const v4f p1 = hv[(kb >> 2) + 1];

    v16bf a;
#pragma unroll
    for (int e = 0; e < 4; ++e) {
        a[e]     = (__bf16)p0[e];
        a[e + 4] = (__bf16)p1[e];
    }
#pragma unroll
    for (int e = 8; e < 16; ++e) a[e] = (__bf16)0.0f;

    const v16bf br = ((const v16bf*)pw_rel)[lane];
    const v16bf bs = ((const v16bf*)pw_root)[lane];

    v8f acc_r = {};
    v8f acc_s = {};
    acc_r = __builtin_amdgcn_wmma_f32_16x16x32_bf16(
        false, a, false, br, (short)0, acc_r, false, false);
    acc_s = __builtin_amdgcn_wmma_f32_16x16x32_bf16(
        false, a, false, bs, (short)0, acc_s, false, false);

#pragma unroll
    for (int j = 0; j < 8; ++j) {
        const int row = row0 + j + half * 8;
        msg [row * C_CH + n] = acc_r[j];
        root[row * C_CH + n] = acc_s[j];
    }
}

// ---------------------------------------------------------------------------
// Edge scatter: agg[dst] += msg[src].  16 threads/edge: 64B coalesced gather
// + 64B of f32 atomics per edge; both 4MB buffers are L2-resident.
// ---------------------------------------------------------------------------
__global__ __launch_bounds__(256) void scatter_kernel(
    const int* __restrict__ src, const int* __restrict__ dst,
    const float* __restrict__ msg, float* __restrict__ agg, int n_edges)
{
    const long long idx = (long long)blockIdx.x * blockDim.x + threadIdx.x;
    const int e = (int)(idx >> 4);
    if (e >= n_edges) return;
    const int c = (int)(idx & 15);
    const int s = src[e];
    const int d = dst[e];
    atomicAdd(&agg[(size_t)d * C_CH + c], msg[(size_t)s * C_CH + c]);
}

// ---------------------------------------------------------------------------
// Fused bias + ReLU + eval-mode BatchNorm.
// ---------------------------------------------------------------------------
__global__ __launch_bounds__(256) void combine_kernel(
    const float* __restrict__ agg, const float* __restrict__ root,
    const float* __restrict__ b,
    const float* __restrict__ g,  const float* __restrict__ be,
    const float* __restrict__ mn, const float* __restrict__ vr,
    float* __restrict__ hout, int total)
{
    const int idx = blockIdx.x * blockDim.x + threadIdx.x;
    if (idx >= total) return;
    const int c = idx & 15;
    float z = agg[idx] + root[idx] + b[c];
    z = fmaxf(z, 0.0f);
    const float scale = g[c] * rsqrtf(vr[c] + 1e-5f);
    hout[idx] = (z - mn[c]) * scale + be[c];
}

// ---------------------------------------------------------------------------
// Per-graph mean pool: one wave per graph; halves merged with ds_swizzle SWAPX16.
// ---------------------------------------------------------------------------
__global__ __launch_bounds__(32) void pool_kernel(
    const float* __restrict__ h, float* __restrict__ pooled)
{
    const int g    = blockIdx.x;
    const int lane = threadIdx.x;
    const int c    = lane & 15;
    const int half = lane >> 4;
    const float* base = h + ((size_t)g * 256 + half * 128) * C_CH + c;
    float s = 0.0f;
#pragma unroll 8
    for (int r = 0; r < 128; ++r) s += base[r * C_CH];
    const int other = __builtin_amdgcn_ds_swizzle(__float_as_int(s), 0x401f); // SWAPX16
    s += __int_as_float(other);
    if (lane < 16) pooled[g * C_CH + c] = s * (1.0f / 256.0f);
}

// ---------------------------------------------------------------------------
// Head MLP: out[g] = relu(pooled[g] @ lw1 + lb1) @ lw2 + lb2.
// ---------------------------------------------------------------------------
__global__ __launch_bounds__(256) void mlp_kernel(
    const float* __restrict__ pooled,
    const float* __restrict__ lw1, const float* __restrict__ lb1,
    const float* __restrict__ lw2, const float* __restrict__ lb2,
    float* __restrict__ out)
{
    const int g = blockIdx.x * blockDim.x + threadIdx.x;
    if (g >= 256) return;
    float p[C_CH];
#pragma unroll
    for (int c = 0; c < C_CH; ++c) p[c] = pooled[g * C_CH + c];
    float o = lb2[0];
#pragma unroll
    for (int j = 0; j < C_CH; ++j) {
        float s = lb1[j];
#pragma unroll
        for (int c = 0; c < C_CH; ++c) s = fmaf(p[c], lw1[c * C_CH + j], s);
        s = fmaxf(s, 0.0f);
        o = fmaf(s, lw2[j], o);
    }
    out[g] = o;
}

extern "C" void kernel_launch(void* const* d_in, const int* in_sizes, int n_in,
                              void* d_out, int out_size, void* d_ws, size_t ws_size,
                              hipStream_t stream)
{
    const float* x    = (const float*)d_in[0];
    const int*   ei   = (const int*)  d_in[1];
    const float* w1r  = (const float*)d_in[2];
    const float* w1s  = (const float*)d_in[3];
    const float* b1   = (const float*)d_in[4];
    const float* w2r  = (const float*)d_in[5];
    const float* w2s  = (const float*)d_in[6];
    const float* b2   = (const float*)d_in[7];
    const float* w3r  = (const float*)d_in[8];
    const float* w3s  = (const float*)d_in[9];
    const float* b3   = (const float*)d_in[10];
    const float* g1   = (const float*)d_in[11];
    const float* be1  = (const float*)d_in[12];
    const float* m1   = (const float*)d_in[13];
    const float* v1   = (const float*)d_in[14];
    const float* g2   = (const float*)d_in[15];
    const float* be2  = (const float*)d_in[16];
    const float* m2   = (const float*)d_in[17];
    const float* v2   = (const float*)d_in[18];
    const float* g3   = (const float*)d_in[19];
    const float* be3  = (const float*)d_in[20];
    const float* m3   = (const float*)d_in[21];
    const float* v3   = (const float*)d_in[22];
    const float* lw1  = (const float*)d_in[23];
    const float* lb1  = (const float*)d_in[24];
    const float* lw2  = (const float*)d_in[25];
    const float* lb2  = (const float*)d_in[26];

    const int* src = ei;
    const int* dst = ei + N_EDGES;

    char* ws = (char*)d_ws;
    const size_t BUF = (size_t)N_NODES * C_CH * sizeof(float);   // 4 MB
    float*  h      = (float*)(ws + 0 * BUF);
    float*  msg    = (float*)(ws + 1 * BUF);
    float*  agg    = (float*)(ws + 2 * BUF);
    float*  root   = (float*)(ws + 3 * BUF);
    float*  pooled = (float*)(ws + 4 * BUF);                     // 16 KB
    __bf16* pw1r   = (__bf16*)(ws + 4 * BUF + 0x10000);          // 8 KB
    __bf16* pw1s   = pw1r + 4096;                                // 8 KB
    __bf16* pw2r   = pw1s + 4096;                                // 1 KB each below
    __bf16* pw2s   = pw2r + 512;
    __bf16* pw3r   = pw2s + 512;
    __bf16* pw3s   = pw3r + 512;

    const int tile_blocks = N_NODES / 16;                        // 4096
    const int edge_blocks = (N_EDGES * 16) / 256;                // 131072
    const int elem_blocks = (N_NODES * C_CH) / 256;              // 4096
    const int total_elems = N_NODES * C_CH;

    // ---- pre-pack weights into WMMA B layout (bf16) ----
    pack_w_kernel<<<16, 256, 0, stream>>>(w1r, pw1r, N_FEAT, 8);
    pack_w_kernel<<<16, 256, 0, stream>>>(w1s, pw1s, N_FEAT, 8);
    pack_w_kernel<<< 2, 256, 0, stream>>>(w2r, pw2r, C_CH, 1);
    pack_w_kernel<<< 2, 256, 0, stream>>>(w2s, pw2s, C_CH, 1);
    pack_w_kernel<<< 2, 256, 0, stream>>>(w3r, pw3r, C_CH, 1);
    pack_w_kernel<<< 2, 256, 0, stream>>>(w3s, pw3s, C_CH, 1);

    // ---- layer 1 ----
    proj1_kernel<<<tile_blocks, 32, 0, stream>>>(x, pw1r, pw1s, msg, root);
    (void)hipMemsetAsync(agg, 0, BUF, stream);
    scatter_kernel<<<edge_blocks, 256, 0, stream>>>(src, dst, msg, agg, N_EDGES);
    combine_kernel<<<elem_blocks, 256, 0, stream>>>(agg, root, b1, g1, be1, m1, v1, h, total_elems);

    // ---- layer 2 ----
    projc_kernel<<<tile_blocks, 32, 0, stream>>>(h, pw2r, pw2s, msg, root);
    (void)hipMemsetAsync(agg, 0, BUF, stream);
    scatter_kernel<<<edge_blocks, 256, 0, stream>>>(src, dst, msg, agg, N_EDGES);
    combine_kernel<<<elem_blocks, 256, 0, stream>>>(agg, root, b2, g2, be2, m2, v2, h, total_elems);

    // ---- layer 3 ----
    projc_kernel<<<tile_blocks, 32, 0, stream>>>(h, pw3r, pw3s, msg, root);
    (void)hipMemsetAsync(agg, 0, BUF, stream);
    scatter_kernel<<<edge_blocks, 256, 0, stream>>>(src, dst, msg, agg, N_EDGES);
    combine_kernel<<<elem_blocks, 256, 0, stream>>>(agg, root, b3, g3, be3, m3, v3, h, total_elems);

    // ---- pool + head ----
    pool_kernel<<<256, 32, 0, stream>>>(h, pooled);
    mlp_kernel<<<1, 256, 0, stream>>>(pooled, lw1, lb1, lw2, lb2, (float*)d_out);
}
